// Head_42331197669540
// MI455X (gfx1250) — compile-verified, matrix operand order checked by reading, not run on hardware
//
#include <hip/hip_runtime.h>
#include <hip/hip_bf16.h>
#include <math.h>

typedef __attribute__((ext_vector_type(16))) _Float16 v16h;
typedef __attribute__((ext_vector_type(8)))  _Float16 v8h;
typedef __attribute__((ext_vector_type(8)))  float    v8f;
typedef __attribute__((ext_vector_type(4)))  float    v4f;

#define B_SZ 512
#define T_SZ 256
#define C_SZ 256
#define H_SZ 64
#define CPAD 264   // 256 + 8 f16 pad
#define HPAD 72    // 64 + 8 f16 pad
#define TPAD 264   // 256 + 8 f16 pad
#define NWAVE 8

struct __align__(16) SMem {
  _Float16 wt[3][64 * CPAD];        // W^T per projection: [h][c], f16
  _Float16 qs[T_SZ * HPAD];         // Q row-major [t][h]
  _Float16 ks[T_SZ * HPAD];         // K row-major [t][h]  (== B-operand of Q K^T)
  _Float16 vts[H_SZ * TPAD];        // V^T [h][t]          (== B-operand of P V)
  _Float16 pscr[NWAVE][16 * TPAD];  // per-wave P relayout scratch [r][t]
  float    bias[3][H_SZ];
};

// Load a 16x32 f16 A/B fragment from a row-major matrix in LDS.
// Per ISA layout: lane holds row/col = lane&15; K chunks {off..off+7} -> v[0..7],
// {off+16..off+23} -> v[8..15], where off already includes (lane>>4)*8.
__device__ __forceinline__ v16h lds_frag(const _Float16* rowbase, int off) {
  v8h lo = *(const v8h*)(rowbase + off);
  v8h hi = *(const v8h*)(rowbase + off + 16);
  v16h r;
#pragma unroll
  for (int i = 0; i < 8; ++i) { r[i] = lo[i]; r[i + 8] = hi[i]; }
  return r;
}

// Same fragment pattern, but sourced from global fp32 with convert-to-f16.
__device__ __forceinline__ v16h gbl_frag_f32(const float* rowbase, int off) {
  v4f a0 = *(const v4f*)(rowbase + off);
  v4f a1 = *(const v4f*)(rowbase + off + 4);
  v4f a2 = *(const v4f*)(rowbase + off + 16);
  v4f a3 = *(const v4f*)(rowbase + off + 20);
  v16h r;
#pragma unroll
  for (int i = 0; i < 4; ++i) {
    r[i]      = (_Float16)a0[i];
    r[i + 4]  = (_Float16)a1[i];
    r[i + 8]  = (_Float16)a2[i];
    r[i + 12] = (_Float16)a3[i];
  }
  return r;
}

__device__ __forceinline__ v8f wmma_f16(v16h a, v16h b, v8f c) {
  return __builtin_amdgcn_wmma_f32_16x16x32_f16(
      /*neg_a=*/false, a, /*neg_b=*/false, b,
      /*c_mod=*/(short)0, c, /*reuse_a=*/false, /*reuse_b=*/false);
}

__global__ __launch_bounds__(256) void fused_attn_head(
    const float* __restrict__ x,
    const float* __restrict__ Wq, const float* __restrict__ bq,
    const float* __restrict__ Wk, const float* __restrict__ bk,
    const float* __restrict__ Wv, const float* __restrict__ bv,
    float* __restrict__ out) {
  __shared__ SMem sm;
  const int tid  = threadIdx.x;
  const int lane = tid & 31;
  const int wid  = tid >> 5;
  const int lh   = lane >> 4;   // which 16-lane half
  const int l16  = lane & 15;
  const int b    = blockIdx.x;

  // Warm L2/L0 with this block's x panel (256 KB) while we stage W below.
  {
    const char* xb = (const char*)(x + (size_t)b * T_SZ * C_SZ);
#pragma unroll
    for (int j = 0; j < 8; ++j)
      __builtin_prefetch(xb + ((size_t)tid + (size_t)j * 256) * 128, 0, 3);
  }

  // ---------------- Phase 0: stage W^T (f16) and biases into LDS ----------------
  for (int idx = tid; idx < 3 * 64 * 256; idx += 256) {
    int p   = idx >> 14;          // / (64*256)
    int rem = idx & 16383;
    int h   = rem >> 8;
    int c   = rem & 255;
    const float* Wp = (p == 0) ? Wq : (p == 1) ? Wk : Wv;
    sm.wt[p][h * CPAD + c] = (_Float16)Wp[c * H_SZ + h];
  }
  if (tid < 192) {
    int p = tid >> 6, h = tid & 63;
    const float* bp = (p == 0) ? bq : (p == 1) ? bk : bv;
    sm.bias[p][h] = bp[h];
  }
  __syncthreads();

  // ---------------- Phase 1: Q/K/V projections (each wave: 2 row-tiles) --------
#pragma unroll 1
  for (int i = 0; i < 2; ++i) {
    const int mt = wid * 2 + i;
    const float* xr = x + ((size_t)b * T_SZ + mt * 16 + l16) * C_SZ;
    v8f acc[3][4];
#pragma unroll
    for (int p = 0; p < 3; ++p)
#pragma unroll
      for (int ht = 0; ht < 4; ++ht) acc[p][ht] = {};

#pragma unroll
    for (int kc = 0; kc < 8; ++kc) {
      const int c0 = kc * 32 + lh * 8;
      v16h a = gbl_frag_f32(xr, c0);   // x tile, read once, cvt f16
#pragma unroll
      for (int p = 0; p < 3; ++p) {
        v16h bm[4];
#pragma unroll
        for (int ht = 0; ht < 4; ++ht)
          bm[ht] = lds_frag(&sm.wt[p][(ht * 16 + l16) * CPAD], c0);
#pragma unroll
        for (int ht = 0; ht < 4; ++ht)
          acc[p][ht] = wmma_f16(a, bm[ht], acc[p][ht]);
      }
    }
    // Epilogue: + bias, cvt f16, scatter into Q / K (row-major) and V^T
#pragma unroll
    for (int p = 0; p < 3; ++p)
#pragma unroll
      for (int ht = 0; ht < 4; ++ht) {
        const float bb = sm.bias[p][ht * 16 + l16];
#pragma unroll
        for (int vr = 0; vr < 8; ++vr) {
          float val = acc[p][ht][vr] + bb;
          int r = mt * 16 + vr + 8 * lh;       // C/D layout: M = vgpr + 8*half
          int h = ht * 16 + l16;               // N = lane&15
          if (p == 0)      sm.qs[r * HPAD + h]  = (_Float16)val;
          else if (p == 1) sm.ks[r * HPAD + h]  = (_Float16)val;
          else             sm.vts[h * TPAD + r] = (_Float16)val;
        }
      }
  }
  __syncthreads();

  // ---------------- Phase 2: causal attention (each wave: its 2 row-tiles) -----
  const float scale = 0.0625f;  // C^-0.5 = 256^-0.5
#pragma unroll 1
  for (int i = 0; i < 2; ++i) {
    const int mt = wid * 2 + i;
    const int r0 = mt * 16;

    // A-operand: this tile's 16 Q rows, K-dim = 64 split into 2 chunks of 32
    v16h aq0 = lds_frag(&sm.qs[(r0 + l16) * HPAD], lh * 8);
    v16h aq1 = lds_frag(&sm.qs[(r0 + l16) * HPAD], 32 + lh * 8);

    // S = Q K^T over all 16 key tiles (kept in registers); 2 key tiles in flight
    v8f sacc[16];
#pragma unroll
    for (int jp = 0; jp < 8; ++jp) {
      const int jt0 = jp * 2, jt1 = jp * 2 + 1;
      v16h b0a = lds_frag(&sm.ks[(jt0 * 16 + l16) * HPAD], lh * 8);
      v16h b1a = lds_frag(&sm.ks[(jt0 * 16 + l16) * HPAD], 32 + lh * 8);
      v16h b0b = lds_frag(&sm.ks[(jt1 * 16 + l16) * HPAD], lh * 8);
      v16h b1b = lds_frag(&sm.ks[(jt1 * 16 + l16) * HPAD], 32 + lh * 8);
      v8f acca = {}, accb = {};
      acca = wmma_f16(aq0, b0a, acca);
      accb = wmma_f16(aq0, b0b, accb);
      acca = wmma_f16(aq1, b1a, acca);
      accb = wmma_f16(aq1, b1b, accb);
      sacc[jt0] = acca;
      sacc[jt1] = accb;
    }

    // Causal mask + softmax (rows live on 16-lane halves -> shfl_xor 1,2,4,8)
    float inv[8];
#pragma unroll
    for (int vr = 0; vr < 8; ++vr) {
      const int r = r0 + vr + 8 * lh;
      float m = -INFINITY;
#pragma unroll
      for (int jt = 0; jt < 16; ++jt) {
        int col = jt * 16 + l16;
        float v = (col <= r) ? sacc[jt][vr] * scale : -INFINITY;
        sacc[jt][vr] = v;
        m = fmaxf(m, v);
      }
      m = fmaxf(m, __shfl_xor(m, 1, 32));
      m = fmaxf(m, __shfl_xor(m, 2, 32));
      m = fmaxf(m, __shfl_xor(m, 4, 32));
      m = fmaxf(m, __shfl_xor(m, 8, 32));
      float s = 0.f;
#pragma unroll
      for (int jt = 0; jt < 16; ++jt) {
        float e = __expf(sacc[jt][vr] - m);   // exp(-inf)=0 handles the mask
        sacc[jt][vr] = e;
        s += e;
      }
      s += __shfl_xor(s, 1, 32);
      s += __shfl_xor(s, 2, 32);
      s += __shfl_xor(s, 4, 32);
      s += __shfl_xor(s, 8, 32);
      inv[vr] = 1.0f / s;
    }

    // Relayout P (C-layout regs) -> per-wave LDS scratch (row-major f16)
    _Float16* pb = &sm.pscr[wid][0];
#pragma unroll
    for (int jt = 0; jt < 16; ++jt)
#pragma unroll
      for (int vr = 0; vr < 8; ++vr)
        pb[(vr + 8 * lh) * TPAD + jt * 16 + l16] =
            (_Float16)(sacc[jt][vr] * inv[vr]);
    __asm volatile("s_wait_dscnt 0" ::: "memory");  // same-wave LDS RAW

    // O = P V  (A = P rows from scratch, B = V^T, K = 256 keys)
    v8f o[4];
#pragma unroll
    for (int ht = 0; ht < 4; ++ht) o[ht] = {};
#pragma unroll
    for (int kc = 0; kc < 8; ++kc) {
      const int k0 = kc * 32 + lh * 8;
      v16h a = lds_frag(&pb[l16 * TPAD], k0);
      v16h bm[4];
#pragma unroll
      for (int ht = 0; ht < 4; ++ht)
        bm[ht] = lds_frag(&sm.vts[(ht * 16 + l16) * TPAD], k0);
#pragma unroll
      for (int ht = 0; ht < 4; ++ht)
        o[ht] = wmma_f16(a, bm[ht], o[ht]);
    }

    // Store fp32 output [b][r][h]
#pragma unroll
    for (int ht = 0; ht < 4; ++ht)
#pragma unroll
      for (int vr = 0; vr < 8; ++vr) {
        int r = r0 + vr + 8 * lh;
        int h = ht * 16 + l16;
        out[((size_t)b * T_SZ + r) * H_SZ + h] = o[ht][vr];
      }
  }
}

extern "C" void kernel_launch(void* const* d_in, const int* in_sizes, int n_in,
                              void* d_out, int out_size, void* d_ws, size_t ws_size,
                              hipStream_t stream) {
  (void)in_sizes; (void)n_in; (void)out_size; (void)d_ws; (void)ws_size;
  const float* x  = (const float*)d_in[0];
  const float* Wq = (const float*)d_in[1];
  const float* bq = (const float*)d_in[2];
  const float* Wk = (const float*)d_in[3];
  const float* bk = (const float*)d_in[4];
  const float* Wv = (const float*)d_in[5];
  const float* bv = (const float*)d_in[6];
  float* out = (float*)d_out;
  fused_attn_head<<<dim3(B_SZ), dim3(256), 0, stream>>>(x, Wq, bq, Wk, bk, Wv, bv, out);
}
